// ReactionDiffusionPDE3D_55576876810627
// MI455X (gfx1250) — matrix operand order repaired and checked by reading, not compiled, not run on hardware
//
#include <hip/hip_runtime.h>

// Reaction-diffusion PDE step, fused single kernel for gfx1250 (MI455X).
//   x: (4, 8, 96, 96, 96) fp32   out: same shape
// - LDS halo tile filled with GLOBAL_LOAD_ASYNC_TO_LDS_B32 (ASYNCcnt path)
// - stencils (laplace + 3 sobels) on VALU from LDS
// - 8x32 channel-mixing einsum via v_wmma_f32_16x16x32_f16 (K = 32 exactly)
// - branchless tanh/sigmoid via v_exp_f32 / v_rcp_f32

#define CDIM 8
#define KD   96
#define TI   16
#define TJ   16
#define TK   4

typedef __attribute__((ext_vector_type(16))) _Float16 v16h;
typedef __attribute__((ext_vector_type(8)))  float    v8f;
typedef __attribute__((ext_vector_type(8)))  unsigned v8u;

static __device__ inline unsigned pkh(float lo, float hi) {
  union { _Float16 h[2]; unsigned u; } t;
  t.h[0] = (_Float16)lo;
  t.h[1] = (_Float16)hi;
  return t.u;
}

#define LOG2E 1.4426950408889634f

// branchless tanh: 1 - 2/(exp2(2*log2e*x)+1); exact at +/-inf saturation
static __device__ inline float fast_tanh(float x) {
  float e = __builtin_amdgcn_exp2f(x * (2.0f * LOG2E));
  return 1.0f - 2.0f * __builtin_amdgcn_rcpf(e + 1.0f);
}

__global__ __launch_bounds__(256)
void rdpde3d_wmma_kernel(const float* __restrict__ x,
                         const float* __restrict__ lmu,
                         const float* __restrict__ ldiff,
                         const float* __restrict__ W,   // (8, 32) row-major fp32
                         float* __restrict__ out) {
  // LDS halo tile: 8 channels x (TK+2) x (TI+2) x (TJ+2) fp32 = 60.75 KB
  __shared__ float xs[CDIM * (TK + 2) * (TI + 2) * (TJ + 2)];

  const int tid  = threadIdx.x;       // 0..255
  const int lane = tid & 31;          // wave32 lane
  const int tj   = tid & 15;
  const int ti   = tid >> 4;          // 0..15

  const int tile = blockIdx.x;        // 0..35  (6x6 tiles of 16x16 over i,j)
  const int i0 = (tile / 6) * TI;
  const int j0 = (tile % 6) * TJ;
  const int k0 = blockIdx.y * TK;     // 0..95 step 4
  const int n  = blockIdx.z;          // 0..3

  const long plane = (long)KD * KD;   // 9216
  const long vol   = plane * KD;      // 884736
  const float* xn = x + (long)n * CDIM * vol;

  // ------- fill LDS halo tile: async global->LDS, zero-pad boundaries -------
  const int NLOAD = CDIM * 6 * 18 * 18;   // 15552
  for (int idx = tid; idx < NLOAD; idx += 256) {
    int c  = idx / (6 * 18 * 18);
    int r  = idx % (6 * 18 * 18);
    int lk = r / (18 * 18);
    int r2 = r % (18 * 18);
    int li = r2 / 18;
    int lj = r2 % 18;
    int gk = k0 + lk - 1, gi = i0 + li - 1, gj = j0 + lj - 1;
    bool inb = (unsigned)gk < (unsigned)KD && (unsigned)gi < (unsigned)KD &&
               (unsigned)gj < (unsigned)KD;
    if (inb) {
      unsigned long long gaddr =
          (unsigned long long)(const void*)(xn + (long)c * vol +
                                            (long)gk * plane + gi * KD + gj);
      unsigned ldsOff = (unsigned)(unsigned long long)(&xs[idx]);
      // async DMA: LDS[ldsOff] = MEM[gaddr]; tracked by ASYNCcnt
      asm volatile("global_load_async_to_lds_b32 %0, %1, off"
                   :: "v"(ldsOff), "v"(gaddr) : "memory");
    } else {
      xs[idx] = 0.0f;   // zero padding (disjoint addresses from async path)
    }
  }
  asm volatile("s_wait_asynccnt 0x0" ::: "memory");

  // ------- A matrix: W zero-padded to 16x32, f16, WMMA A-layout -------
  // A 16x32 as packed here (contiguous k, same order as B): lanes 0-15 row M=lane hold K=0..15 (v0-7)
  //                        lanes 16-31 row M=lane-16 hold K=16..31
  const int arow = lane & 15;
  const int kblo = (lane < 16) ? 0 : 16;
  v16h a;
  if (arow < CDIM) {
    const float* wr = W + arow * 32;
#pragma unroll
    for (int t = 0; t < 8; ++t) {
      int k = kblo + 2 * t;
      a[2 * t]     = (_Float16)wr[k];
      a[2 * t + 1] = (_Float16)wr[k + 1];
    }
  } else {
#pragma unroll
    for (int t = 0; t < 16; ++t) a[t] = (_Float16)0.0f;
  }

  // mu = sigmoid(lmu); dcoef = exp(-3 + ldiff)  (branchless exp2/rcp forms)
  const float mu =
      __builtin_amdgcn_rcpf(1.0f + __builtin_amdgcn_exp2f(-lmu[0] * LOG2E));
  const float dcoef = __builtin_amdgcn_exp2f((ldiff[0] - 3.0f) * LOG2E);
  const float omu   = 1.0f - mu;

  __syncthreads();

  const int gi = i0 + ti, gj = j0 + tj;

  for (int tk = 0; tk < TK; ++tk) {
    float feat[32];
    float lap[CDIM], xc[CDIM];

#pragma unroll
    for (int c = 0; c < CDIM; ++c) {
      const float* s = &xs[(((c * 6) + tk) * 18 + ti) * 18 + tj];
      float v[27];
#pragma unroll
      for (int dz = 0; dz < 3; ++dz)
#pragma unroll
        for (int dy = 0; dy < 3; ++dy)
#pragma unroll
          for (int dx = 0; dx < 3; ++dx)
            v[dz * 9 + dy * 3 + dx] = s[dz * 324 + dy * 18 + dx];

      const float v111 = v[13];
      const float lapc = (v[4] + v[22] + v[10] + v[16] + v[12] + v[14]
                          - 6.0f * v111) * (1.0f / 6.0f);
      float gx = 0.0f, gy = 0.0f, gz = 0.0f;
#pragma unroll
      for (int dz = 0; dz < 3; ++dz) {
        const float zw = (dz == 1) ? 2.0f : 1.0f;
#pragma unroll
        for (int d = 0; d < 3; ++d) {
          const float w = zw * ((d == 1) ? 2.0f : 1.0f);
          gx += w * (v[dz * 9 + d * 3 + 0] - v[dz * 9 + d * 3 + 2]);
          gy += w * (v[dz * 9 + 0 * 3 + d] - v[dz * 9 + 2 * 3 + d]);
        }
      }
#pragma unroll
      for (int dy = 0; dy < 3; ++dy) {
        const float yw = (dy == 1) ? 2.0f : 1.0f;
#pragma unroll
        for (int dx = 0; dx < 3; ++dx) {
          const float w = yw * ((dx == 1) ? 2.0f : 1.0f);
          gz += w * (v[0 * 9 + dy * 3 + dx] - v[2 * 9 + dy * 3 + dx]);
        }
      }
      gx *= 0.0625f; gy *= 0.0625f; gz *= 0.0625f;

      xc[c]  = v111;
      lap[c] = lapc;
      feat[c]             = v111;  // feat[0..7]  = x channels
      feat[8 + 3 * c + 0] = gz;    // state order: laplace, sobel_z, sobel_x, sobel_y
      feat[8 + 3 * c + 1] = gx;
      feat[8 + 3 * c + 2] = gy;
    }

    // pack feat (K=0..31) into 16 half2 words; P[r] = (feat[2r], feat[2r+1])
    unsigned P[16];
#pragma unroll
    for (int r = 0; r < 16; ++r) P[r] = pkh(feat[2 * r], feat[2 * r + 1]);

    // B 32x16 f16 as packed here (contiguous k, as in A): lanes 0-15 hold K=0..15 of column N=lane,
    // lanes 16-31 hold K=16..31 of column N=lane-16.  One WMMA covers the
    // 16 points owned by lanes 0-15 (b0), a second covers lanes 16-31 (b1).
    v8u b0u, b1u;
#pragma unroll
    for (int r = 0; r < 8; ++r) {
      unsigned fromLow  = __shfl(P[r + 8], lane & 15, 32);        // lane-16's hi half
      unsigned fromHigh = __shfl(P[r], (lane & 15) + 16, 32);     // lane+16's lo half
      b0u[r] = (lane < 16) ? P[r]     : fromLow;
      b1u[r] = (lane < 16) ? fromHigh : P[r + 8];
    }
    union { v8u u; v16h h; } cb0, cb1;
    cb0.u = b0u; cb1.u = b1u;

    v8f c0 = {0.f, 0.f, 0.f, 0.f, 0.f, 0.f, 0.f, 0.f};
    v8f d0 = __builtin_amdgcn_wmma_f32_16x16x32_f16(
        false, a, false, cb0.h, (short)0, c0, false, false);
    v8f d1 = __builtin_amdgcn_wmma_f32_16x16x32_f16(
        false, a, false, cb1.h, (short)0, c0, false, false);

    // D layout: VGPR r, lanes 0-15 = (row r, col lane).  d0 col n = point of
    // lane n; d1 col n = point of lane n+16 -> shuffle back to owner.
    float react[CDIM];
#pragma unroll
    for (int r = 0; r < CDIM; ++r) {
      float hi = __shfl(d1[r], lane & 15, 32);
      react[r] = (lane < 16) ? d0[r] : hi;
    }

    // ------- combine + store -------
    const int gk = k0 + tk;
    float* on = out + (long)n * CDIM * vol + (long)gk * plane + gi * KD + gj;
#pragma unroll
    for (int c = 0; c < CDIM; ++c) {
      float val = xc[c] + mu * (dcoef * lap[c]) + omu * fast_tanh(react[c]);
      on[(long)c * vol] = fmaxf(val, 0.0f);
    }
  }
}

extern "C" void kernel_launch(void* const* d_in, const int* in_sizes, int n_in,
                              void* d_out, int out_size, void* d_ws, size_t ws_size,
                              hipStream_t stream) {
  const float* x      = (const float*)d_in[0];
  // d_in[1] = stencil kernel: fixed values, baked into the kernel.
  const float* lmu    = (const float*)d_in[2];
  const float* ldiff  = (const float*)d_in[3];
  const float* W      = (const float*)d_in[4];
  float* out          = (float*)d_out;

  dim3 grid(36, 24, 4);   // (i,j) tiles, k chunks, batch
  dim3 block(256);
  hipLaunchKernelGGL(rdpde3d_wmma_kernel, grid, block, 0, stream,
                     x, lmu, ldiff, W, out);
}
